// SioConvLayer_6322191860142
// MI455X (gfx1250) — matrix operand order repaired and checked by reading, not compiled, not run on hardware
//
#include <hip/hip_runtime.h>

typedef _Float16 half_t;
typedef __attribute__((ext_vector_type(16))) _Float16 v16h;
typedef __attribute__((ext_vector_type(8)))  _Float16 v8h;
typedef __attribute__((ext_vector_type(4)))  _Float16 v4h;
typedef __attribute__((ext_vector_type(8)))  float    v8f;

#define Bq   4
#define Lq   2048
#define DIMq 1024
#define FFq  4096
#define Hq   8
#define DIq  64
#define DDq  64
#define Mq   (Bq*Lq)        /* 8192 rows */
#define NCH  (Bq*Hq*DDq)    /* 2048 scan channels */
#define NCHUNK 32
#define CLEN (Lq/NCHUNK)    /* 64 */

#define EP_F32    1
#define EP_SILU16 2
#define EP_RES    4

// issue one 32-K LDS tile (A:16B x2 + B:16B x2 per thread) via CDNA5 async copy.
// GVS mode: addr = SGPR base + VGPR 32-bit offset (+ inst offset, applied to BOTH
// the global address and the LDS address).
#define ASYNC_TILE(lA, lB, gofs, Ab, Bb)                                   \
  asm volatile(                                                            \
      "global_load_async_to_lds_b128 %0, %2, %3\n\t"                       \
      "global_load_async_to_lds_b128 %0, %2, %3 offset:16\n\t"             \
      "global_load_async_to_lds_b128 %1, %2, %4\n\t"                       \
      "global_load_async_to_lds_b128 %1, %2, %4 offset:16"                 \
      :: "v"(lA), "v"(lB), "v"(gofs), "s"(Ab), "s"(Bb) : "memory")

// ---------------- fp32 -> fp16 conversion (4-wide) ----------------
__global__ void k_cvt(const float4* __restrict__ s, v4h* __restrict__ d, int n4) {
  int i = blockIdx.x * blockDim.x + threadIdx.x;
  if (i < n4) {
    float4 f = s[i];
    v4h o;
    o.x = (half_t)f.x; o.y = (half_t)f.y; o.z = (half_t)f.z; o.w = (half_t)f.w;
    d[i] = o;
  }
}

// ---------------- WMMA GEMM: C[M,N] = A[M,K] * B[N,K]^T + bias ----------------
// A, B fp16 row-major (K contiguous).  Async double-buffered LDS pipeline.
__global__ __launch_bounds__(256) void k_gemm(
    const half_t* __restrict__ A, const half_t* __restrict__ Bw,
    const float* __restrict__ bias, const float* __restrict__ Res,
    float* __restrict__ C32, half_t* __restrict__ C16,
    int M, int N, int K, int flags)
{
  __shared__ half_t As[2][128 * 40];   // pitch 40 halves (80B) -> conflict-free b128
  __shared__ half_t Bs[2][128 * 40];

  const int tid  = threadIdx.x;
  const int lane = tid & 31;
  const int wv   = tid >> 5;
  const int wm   = wv & 3;          // 4 wave-rows of 32
  const int wn   = wv >> 2;         // 2 wave-cols of 64
  const int bm   = blockIdx.y, bn = blockIdx.x;

  v8f acc[2][4];
#pragma unroll
  for (int i = 0; i < 2; ++i)
#pragma unroll
    for (int j = 0; j < 4; ++j)
#pragma unroll
      for (int r = 0; r < 8; ++r) acc[i][j][r] = 0.0f;

  // async staging: each thread owns one 32-byte row segment of each tile
  const int lrow = tid >> 1;
  const int lks  = (tid & 1) << 4;                 // 0 or 16 halves
  const half_t* Abase = A  + (size_t)(bm * 128) * K;
  const half_t* Bbase = Bw + (size_t)(bn * 128) * K;
  const unsigned goff = (unsigned)((lrow * K + lks) * sizeof(half_t));
  const unsigned ldsA0 = (unsigned)(size_t)&As[0][lrow * 40 + lks];
  const unsigned ldsA1 = (unsigned)(size_t)&As[1][lrow * 40 + lks];
  const unsigned ldsB0 = (unsigned)(size_t)&Bs[0][lrow * 40 + lks];
  const unsigned ldsB1 = (unsigned)(size_t)&Bs[1][lrow * 40 + lks];

  // fragment read bases (ISA VGPR layouts, wave32)
  const int frow = lane & 15;
  const int khal = lane >> 4;                      // half-wave selector
  const int aoff = (wm * 32 + frow) * 40 + khal * 8;   // A: K {k..k+7, k+16..k+23}
  const int boff = (wn * 64 + frow) * 40 + khal * 8;   // B: K {k..k+7, k+16..k+23}
  const half_t* AsR0 = &As[0][aoff];
  const half_t* AsR1 = &As[1][aoff];
  const half_t* BsR0 = &Bs[0][boff];
  const half_t* BsR1 = &Bs[1][boff];

  const int steps = K >> 5;
  ASYNC_TILE(ldsA0, ldsB0, goff, Abase, Bbase);    // prefetch tile 0

  for (int kt = 0; kt < steps; ++kt) {
    asm volatile("s_wait_asynccnt 0x0" ::: "memory");  // own tile-kt copies done
    __syncthreads();                                    // everyone's done; prev reads done

    if (kt + 1 < steps) {                               // prefetch tile kt+1
      const half_t* Ak = Abase + (kt + 1) * 32;
      const half_t* Bk = Bbase + (kt + 1) * 32;
      if (kt & 1) { ASYNC_TILE(ldsA0, ldsB0, goff, Ak, Bk); }
      else        { ASYNC_TILE(ldsA1, ldsB1, goff, Ak, Bk); }
    }

    const half_t* AsR = (kt & 1) ? AsR1 : AsR0;
    const half_t* BsR = (kt & 1) ? BsR1 : BsR0;

    v16h af[2], bf[4];
#pragma unroll
    for (int i = 0; i < 2; ++i) {
      union { v16h v; v8h h[2]; } t;
      const half_t* p = AsR + i * 16 * 40;
      t.h[0] = *(const v8h*)p;
      t.h[1] = *(const v8h*)(p + 16);
      af[i] = t.v;
    }
#pragma unroll
    for (int j = 0; j < 4; ++j) {
      union { v16h v; v8h h[2]; } t;
      const half_t* p = BsR + j * 16 * 40;
      t.h[0] = *(const v8h*)p;
      t.h[1] = *(const v8h*)(p + 16);
      bf[j] = t.v;
    }
#pragma unroll
    for (int i = 0; i < 2; ++i)
#pragma unroll
      for (int j = 0; j < 4; ++j)
        acc[i][j] = __builtin_amdgcn_wmma_f32_16x16x32_f16(
            false, af[i], false, bf[j], (short)0, acc[i][j], false, false);
  }

  const int row0 = bm * 128 + wm * 32;
  const int col0 = bn * 128 + wn * 64;
#pragma unroll
  for (int i = 0; i < 2; ++i) {
#pragma unroll
    for (int j = 0; j < 4; ++j) {
      const int n  = col0 + j * 16 + (lane & 15);
      const float bv = bias[n];
      const int m0 = row0 + i * 16 + khal * 8;
#pragma unroll
      for (int r = 0; r < 8; ++r) {
        float v = acc[i][j][r] + bv;
        const size_t idx = (size_t)(m0 + r) * N + n;
        if (flags & EP_RES) v += Res[idx];
        if (flags & EP_F32) C32[idx] = v;
        if (flags & EP_SILU16) {
          const float sv = v / (1.0f + __expf(-v));
          C16[idx] = (half_t)sv;
        }
      }
    }
  }
}

// ---------------- per-(b,l): a-normalization + vx = mw @ xc ----------------
__global__ __launch_bounds__(256) void k_prep(
    const float* __restrict__ xz, const float* __restrict__ az,
    const float* __restrict__ matw,
    float* __restrict__ a_out, float* __restrict__ v_out)
{
  __shared__ float xs[1024];
  const int row = blockIdx.x;               // b*L + l
  const int tid = threadIdx.x;
  const float* xrow = xz + (size_t)row * 1024;
  for (int i = tid; i < 1024; i += 256) xs[i] = xrow[i];
  __syncthreads();

  const float* arow = az + (size_t)row * 1024;
  float* aout = a_out + (size_t)row * 1024;
  float* vout = v_out + (size_t)row * 1024;

  for (int idx = tid; idx < 512; idx += 256) {
    const int h = idx >> 6, e = idx & 63;
    // scale = rsqrt(m2)*sigmoid(log(m2)) == sqrt(m2)/(1+m2)
    const float ar = arow[idx * 2], ai = arow[idx * 2 + 1];
    const float m2 = ar * ar + ai * ai;
    const float s  = sqrtf(m2) / (1.0f + m2);
    aout[idx * 2]     = ar * s;
    aout[idx * 2 + 1] = ai * s;

    const float* wrow = matw + ((size_t)(h * 64 + e) * 64) * 2;
    const float* xh   = xs + h * 128;
    float vr = 0.f, vi = 0.f;
#pragma unroll 8
    for (int d = 0; d < 64; ++d) {
      const float wr = wrow[d * 2], wi = wrow[d * 2 + 1];
      const float xr = xh[d * 2],  xi = xh[d * 2 + 1];
      vr = fmaf(wr, xr, fmaf(-wi, xi, vr));
      vi = fmaf(wr, xi, fmaf( wi, xr, vi));
    }
    vout[idx * 2]     = vr;
    vout[idx * 2 + 1] = vi;
  }
}

// ---------------- chunked scan, pass 1: per-(channel,chunk) affine aggregate ----------------
__global__ void k_scan_agg(const float* __restrict__ a_c, const float* __restrict__ v_c,
                           float* __restrict__ agg)
{
  const int g  = blockIdx.x * blockDim.x + threadIdx.x;   // 65536
  const int c  = g & (NCH - 1);
  const int j  = g >> 11;
  const int b  = c >> 9;
  const int hd = c & 511;
  float Ar = 1.f, Ai = 0.f, Br = 0.f, Bi = 0.f;
  const int t0 = j * CLEN;
#pragma unroll 4
  for (int t = 0; t < CLEN; ++t) {
    const size_t base = ((size_t)(b * Lq + t0 + t) * 512 + hd) * 2;
    const float ar = a_c[base], ai = a_c[base + 1];
    const float vr = v_c[base], vi = v_c[base + 1];
    const float nAr = ar * Ar - ai * Ai;
    const float nAi = ar * Ai + ai * Ar;
    const float nBr = ar * Br - ai * Bi + vr;
    const float nBi = ar * Bi + ai * Br + vi;
    Ar = nAr; Ai = nAi; Br = nBr; Bi = nBi;
  }
  float* o = agg + (size_t)(c * NCHUNK + j) * 4;
  o[0] = Ar; o[1] = Ai; o[2] = Br; o[3] = Bi;
}

// ---------------- pass 2: per-channel chunk prefix; emits hidden_next ----------------
__global__ void k_scan_prefix(const float* __restrict__ agg, const float* __restrict__ h0,
                              float* __restrict__ ent, float* __restrict__ hid_out)
{
  const int c = blockIdx.x * blockDim.x + threadIdx.x;    // 2048
  float hr = h0[c * 2], hi = h0[c * 2 + 1];
  for (int j = 0; j < NCHUNK; ++j) {
    float* e = ent + (size_t)(c * NCHUNK + j) * 2;
    e[0] = hr; e[1] = hi;
    const float* gg = agg + (size_t)(c * NCHUNK + j) * 4;
    const float Ar = gg[0], Ai = gg[1], Br = gg[2], Bi = gg[3];
    const float nr = Ar * hr - Ai * hi + Br;
    const float ni = Ar * hi + Ai * hr + Bi;
    hr = nr; hi = ni;
  }
  hid_out[c * 2]     = hr;
  hid_out[c * 2 + 1] = hi;
}

// ---------------- pass 3: re-emit full hidden sequence ----------------
__global__ void k_scan_emit(const float* __restrict__ a_c, const float* __restrict__ v_c,
                            const float* __restrict__ ent, float* __restrict__ hs)
{
  const int g  = blockIdx.x * blockDim.x + threadIdx.x;
  const int c  = g & (NCH - 1);
  const int j  = g >> 11;
  const int b  = c >> 9;
  const int hd = c & 511;
  float hr = ent[(size_t)(c * NCHUNK + j) * 2];
  float hi = ent[(size_t)(c * NCHUNK + j) * 2 + 1];
  const int t0 = j * CLEN;
#pragma unroll 4
  for (int t = 0; t < CLEN; ++t) {
    const size_t base = ((size_t)(b * Lq + t0 + t) * 512 + hd) * 2;
    const float ar = a_c[base], ai = a_c[base + 1];
    const float vr = v_c[base], vi = v_c[base + 1];
    const float nr = ar * hr - ai * hi + vr;
    const float ni = ar * hi + ai * hr + vi;
    hr = nr; hi = ni;
    hs[base]     = hr;
    hs[base + 1] = hi;
  }
}

// ---------------- per-(b,l): hout = mv @ hseq, interleave -> hr (fp16) ----------------
__global__ __launch_bounds__(256) void k_hout(
    const float* __restrict__ hs, const float* __restrict__ matv,
    half_t* __restrict__ hr_out)
{
  __shared__ float hsl[1024];
  const int row = blockIdx.x;
  const int tid = threadIdx.x;
  const float* hrow = hs + (size_t)row * 1024;
  for (int i = tid; i < 1024; i += 256) hsl[i] = hrow[i];
  __syncthreads();

  half_t* orow = hr_out + (size_t)row * 1024;
  for (int idx = tid; idx < 512; idx += 256) {
    const int h = idx >> 6, n = idx & 63;
    const float* wrow = matv + ((size_t)(h * 64 + n) * 64) * 2;
    const float* hh   = hsl + h * 128;
    float vr = 0.f, vi = 0.f;
#pragma unroll 8
    for (int o = 0; o < 64; ++o) {
      const float wr = wrow[o * 2], wi = wrow[o * 2 + 1];
      const float xr = hh[o * 2],  xi = hh[o * 2 + 1];
      vr = fmaf(wr, xr, fmaf(-wi, xi, vr));
      vi = fmaf(wr, xi, fmaf( wi, xr, vi));
    }
    orow[idx * 2]     = (half_t)vr;
    orow[idx * 2 + 1] = (half_t)vi;
  }
}

// ---------------- host side ----------------
extern "C" void kernel_launch(void* const* d_in, const int* in_sizes, int n_in,
                              void* d_out, int out_size, void* d_ws, size_t ws_size,
                              hipStream_t stream)
{
  (void)in_sizes; (void)n_in; (void)out_size; (void)ws_size;
  const float* x      = (const float*)d_in[0];
  const float* hidden = (const float*)d_in[1];
  const float* w_in   = (const float*)d_in[2];
  const float* b_in   = (const float*)d_in[3];
  const float* w_x    = (const float*)d_in[4];
  const float* b_x    = (const float*)d_in[5];
  const float* w_a    = (const float*)d_in[6];
  const float* b_a    = (const float*)d_in[7];
  const float* w_o1   = (const float*)d_in[8];
  const float* b_o1   = (const float*)d_in[9];
  const float* w_o2   = (const float*)d_in[10];
  const float* b_o2   = (const float*)d_in[11];
  const float* mat_v  = (const float*)d_in[12];
  const float* mat_w  = (const float*)d_in[13];
  float* out = (float*)d_out;   // y (Mq*DIMq) then hidden_next (NCH*2)

  char* wsp = (char*)d_ws;
  size_t off = 0;
  auto alloc = [&](size_t bytes) -> char* {
    char* p = wsp + off;
    off = (off + bytes + 255) & ~(size_t)255;
    return p;
  };

  half_t* xh    = (half_t*)alloc((size_t)Mq * DIMq * 2);
  half_t* winh  = (half_t*)alloc((size_t)FFq * DIMq * 2);
  half_t* wxh   = (half_t*)alloc((size_t)1024 * FFq * 2);
  half_t* wah   = (half_t*)alloc((size_t)1024 * FFq * 2);
  half_t* wo1h  = (half_t*)alloc((size_t)FFq * 1024 * 2);
  half_t* wo2h  = (half_t*)alloc((size_t)DIMq * FFq * 2);
  float*  u     = (float*)alloc((size_t)Mq * FFq * 4);
  half_t* uah   = (half_t*)alloc((size_t)Mq * FFq * 2);
  float*  xz    = (float*)alloc((size_t)Mq * 1024 * 4);
  float*  az    = (float*)alloc((size_t)Mq * 1024 * 4);
  float*  a_c   = (float*)alloc((size_t)Mq * 1024 * 4);
  float*  v_c   = (float*)alloc((size_t)Mq * 1024 * 4);
  float*  hs    = (float*)alloc((size_t)Mq * 1024 * 4);
  half_t* hrh   = (half_t*)alloc((size_t)Mq * 1024 * 2);
  half_t* y1h   = (half_t*)alloc((size_t)Mq * FFq * 2);
  float*  agg   = (float*)alloc((size_t)NCH * NCHUNK * 4 * 4);
  float*  ent   = (float*)alloc((size_t)NCH * NCHUNK * 2 * 4);

  // fp32 -> fp16 conversions
  auto cvt = [&](const float* s, half_t* d, size_t n) {
    int n4 = (int)(n / 4);
    k_cvt<<<(n4 + 255) / 256, 256, 0, stream>>>((const float4*)s, (v4h*)d, n4);
  };
  cvt(x,    xh,   (size_t)Mq * DIMq);
  cvt(w_in, winh, (size_t)FFq * DIMq);
  cvt(w_x,  wxh,  (size_t)1024 * FFq);
  cvt(w_a,  wah,  (size_t)1024 * FFq);
  cvt(w_o1, wo1h, (size_t)FFq * 1024);
  cvt(w_o2, wo2h, (size_t)DIMq * FFq);

  // fc_in: u = x@W^T + b  (store u f32 for residual, silu(u) f16 for next GEMMs)
  {
    dim3 g(FFq / 128, Mq / 128);
    k_gemm<<<g, 256, 0, stream>>>(xh, winh, b_in, nullptr, u, uah,
                                  Mq, FFq, DIMq, EP_F32 | EP_SILU16);
  }
  // fc_x / fc_a
  {
    dim3 g(1024 / 128, Mq / 128);
    k_gemm<<<g, 256, 0, stream>>>(uah, wxh, b_x, nullptr, xz, nullptr,
                                  Mq, 1024, FFq, EP_F32);
    k_gemm<<<g, 256, 0, stream>>>(uah, wah, b_a, nullptr, az, nullptr,
                                  Mq, 1024, FFq, EP_F32);
  }
  // a-normalization + vx
  k_prep<<<Mq, 256, 0, stream>>>(xz, az, mat_w, a_c, v_c);

  // chunked scan
  k_scan_agg<<<(NCH * NCHUNK) / 256, 256, 0, stream>>>(a_c, v_c, agg);
  k_scan_prefix<<<NCH / 256, 256, 0, stream>>>(agg, hidden, ent,
                                               out + (size_t)Mq * DIMq);
  k_scan_emit<<<(NCH * NCHUNK) / 256, 256, 0, stream>>>(a_c, v_c, ent, hs);

  // hout = mv @ hseq  -> hr (fp16)
  k_hout<<<Mq, 256, 0, stream>>>(hs, mat_v, hrh);

  // fc_out1: y1 = silu(hr@W^T + b + u)
  {
    dim3 g(FFq / 128, Mq / 128);
    k_gemm<<<g, 256, 0, stream>>>(hrh, wo1h, b_o1, u, nullptr, y1h,
                                  Mq, FFq, 1024, EP_RES | EP_SILU16);
  }
  // fc_out2: y = y1@W^T + b  -> d_out
  {
    dim3 g(DIMq / 128, Mq / 128);
    k_gemm<<<g, 256, 0, stream>>>(y1h, wo2h, b_o2, nullptr, out, nullptr,
                                  Mq, DIMq, FFq, EP_F32);
  }
}
